// GatingMixedDecoder_25022479466640
// MI455X (gfx1250) — matrix operand order repaired and checked by reading, not compiled, not run on hardware
//
#include <hip/hip_runtime.h>
#include <hip/hip_bf16.h>
#include <math.h>

typedef _Float16 v16h __attribute__((ext_vector_type(16)));
typedef _Float16 v8h  __attribute__((ext_vector_type(8)));
typedef float    v8f  __attribute__((ext_vector_type(8)));

union Frag { v16h v; v8h h[2]; };

#define B_ROWS   8192
#define CHUNK    4096

__device__ __forceinline__ float elu1(float x) {
  return x > 0.f ? x : (__expf(x) - 1.f);
}

// ---------------------------------------------------------------------------
// Kernel 1: gating MLP + softmax -> coeff [B, 8]; one block (128 thr) per row
// ---------------------------------------------------------------------------
__global__ __launch_bounds__(128)
void gate_kernel(const float* __restrict__ z, const float* __restrict__ c,
                 const float* __restrict__ g0w, const float* __restrict__ g0b,
                 const float* __restrict__ g1w, const float* __restrict__ g1b,
                 const float* __restrict__ g2w, const float* __restrict__ g2b,
                 float* __restrict__ coeff)
{
  __shared__ float sx[320];
  __shared__ float sh[128];
  const int b = blockIdx.x;
  const int t = threadIdx.x;

  for (int j = t; j < 320; j += 128)
    sx[j] = (j < 64) ? z[b * 64 + j] : c[(size_t)b * 256 + (j - 64)];
  __syncthreads();

  float a0 = g0b[t];
  #pragma unroll 4
  for (int j = 0; j < 320; ++j) a0 += sx[j] * g0w[j * 128 + t];
  a0 = elu1(a0);
  sh[t] = a0;
  __syncthreads();

  float a1 = g1b[t];
  #pragma unroll 4
  for (int j = 0; j < 128; ++j) a1 += sh[j] * g1w[j * 128 + t];
  a1 = elu1(a1);
  __syncthreads();
  sh[t] = a1;
  __syncthreads();

  if (t < 8) {
    float l = g2b[t];
    #pragma unroll 4
    for (int j = 0; j < 128; ++j) l += sh[j] * g2w[j * 8 + t];
    sx[t] = l;
  }
  __syncthreads();
  if (t == 0) {
    float mx = sx[0];
    for (int e = 1; e < 8; ++e) mx = fmaxf(mx, sx[e]);
    float ex[8]; float s = 0.f;
    for (int e = 0; e < 8; ++e) { ex[e] = __expf(sx[e] - mx); s += ex[e]; }
    float inv = 1.f / s;
    for (int e = 0; e < 8; ++e) coeff[b * 8 + e] = ex[e] * inv;
  }
}

// ---------------------------------------------------------------------------
// Kernel 2: stack expert weights + bias rows, TRANSPOSED to f16 [N][Kd].
// Wt[n*Kd + r] = r<8J ? w[r*N+n] : (r<8J+8 ? bias[(r-8J)*N+n] : 0)
// Tiled 32x32 LDS transpose; grid (Kd/32, N/32), block 256
// ---------------------------------------------------------------------------
__global__ __launch_bounds__(256)
void convert_w_t(const float* __restrict__ w, const float* __restrict__ bias,
                 _Float16* __restrict__ Wt, int rows8J, int Kd, int N)
{
  __shared__ float tile[32][33];
  const int rb = blockIdx.x * 32;     // Kd-dim base
  const int nb = blockIdx.y * 32;     // N-dim base
  const int tx = threadIdx.x & 31;
  const int ty = threadIdx.x >> 5;    // 0..7

  #pragma unroll
  for (int i = ty; i < 32; i += 8) {
    int r = rb + i, n = nb + tx;
    float v;
    if (r < rows8J)           v = w[(size_t)r * N + n];
    else if (r < rows8J + 8)  v = bias[(size_t)(r - rows8J) * N + n];
    else                      v = 0.f;
    tile[i][tx] = v;
  }
  __syncthreads();
  #pragma unroll
  for (int i = ty; i < 32; i += 8) {
    int n = nb + i, r = rb + tx;
    Wt[(size_t)n * Kd + r] = (_Float16)tile[tx][i];
  }
}

// ---------------------------------------------------------------------------
// Kernel 3: LN(concat(z, prev)) then 8 coeff-scaled f16 copies + coeff tail
// -> A[localrow][Kd].  One block (256 thr) per row.
// ---------------------------------------------------------------------------
__global__ __launch_bounds__(256)
void ln_scale(const float* __restrict__ z, const float* __restrict__ prev,
              const float* __restrict__ coeff, _Float16* __restrict__ A,
              int row0, int P, int J, int Kd)
{
  __shared__ float sx[1088];
  __shared__ float r1[256];
  __shared__ float r2[256];
  __shared__ float cf[8];
  const int t = threadIdx.x;
  const int row = row0 + blockIdx.x;

  float s = 0.f, s2 = 0.f;
  for (int j = t; j < J; j += 256) {
    float v = (j < 64) ? z[row * 64 + j] : prev[(size_t)row * P + (j - 64)];
    sx[j] = v; s += v; s2 += v * v;
  }
  r1[t] = s; r2[t] = s2;
  if (t < 8) cf[t] = coeff[row * 8 + t];
  __syncthreads();
  #pragma unroll
  for (int off = 128; off > 0; off >>= 1) {
    if (t < off) { r1[t] += r1[t + off]; r2[t] += r2[t + off]; }
    __syncthreads();
  }
  const float mean = r1[0] / (float)J;
  const float var  = r2[0] / (float)J - mean * mean;
  const float rstd = rsqrtf(var + 1e-5f);

  _Float16* dst = A + (size_t)blockIdx.x * Kd;
  #pragma unroll
  for (int e = 0; e < 8; ++e) {
    const float ce = cf[e];
    for (int j = t; j < J; j += 256)
      dst[e * J + j] = (_Float16)(ce * (sx[j] - mean) * rstd);
  }
  if (t < 32)
    dst[8 * J + t] = (t < 8) ? (_Float16)cf[t] : (_Float16)0.f;
}

// ---------------------------------------------------------------------------
// Kernel 4: WMMA GEMM.  C[M=CHUNK, N] = A[M, Kd] * Wt[N, Kd]^T (f16, f32 acc)
// 128x256 block tile, 8 waves as 2(M)x4(N); wave tile 64x64 = 4x4 WMMA.
// K-step 32, double-buffered LDS with register staging.
// ---------------------------------------------------------------------------
__global__ __launch_bounds__(256)
void gemm_moe(const _Float16* __restrict__ A, const _Float16* __restrict__ Wt,
              float* __restrict__ Out, int Kd, int N, int ldc, int row0,
              int do_elu)
{
  __shared__ _Float16 As[2][128][40];   // [buf][m][k]
  __shared__ _Float16 Bs[2][256][40];   // [buf][n][k]

  const int tid  = threadIdx.x;
  const int lane = tid & 31;
  const int wv   = tid >> 5;
  const int wm   = wv >> 2;           // 2 M-groups of 64 rows
  const int wn   = wv & 3;            // 4 N-groups of 64 cols
  const int mblock = blockIdx.y * 128;
  const int nbase  = blockIdx.x * 256;
  const int lr   = lane & 15;
  const int ka   = (lane >> 4) * 16;  // A-frag K sub-offset (halves)
  const int kb   = (lane >> 4) * 16;  // B-frag K sub-offset (halves)

  // staging: thread owns contiguous 32B slices (A: 1 row-slice, B: 2)
  const int srow = tid >> 1;              // 0..127
  const int soff = (tid & 1) * 16;        // 0 or 16 halves
  const _Float16* pa  = A  + (size_t)(mblock + srow) * Kd + soff;
  const _Float16* pb0 = Wt + (size_t)(nbase + srow) * Kd + soff;
  const _Float16* pb1 = Wt + (size_t)(nbase + srow + 128) * Kd + soff;

  v8f acc[4][4] = {};
  v8h av0, av1, b00, b01, b10, b11;

  // prologue: tile 0
  av0 = *(const v8h*)pa;        av1 = *(const v8h*)(pa + 8);
  b00 = *(const v8h*)pb0;       b01 = *(const v8h*)(pb0 + 8);
  b10 = *(const v8h*)pb1;       b11 = *(const v8h*)(pb1 + 8);
  pa += 32; pb0 += 32; pb1 += 32;
  *(v8h*)&As[0][srow][soff]           = av0;
  *(v8h*)&As[0][srow][soff + 8]       = av1;
  *(v8h*)&Bs[0][srow][soff]           = b00;
  *(v8h*)&Bs[0][srow][soff + 8]       = b01;
  *(v8h*)&Bs[0][srow + 128][soff]     = b10;
  *(v8h*)&Bs[0][srow + 128][soff + 8] = b11;
  __syncthreads();

  int buf = 0;
  for (int kk = 0; kk < Kd; kk += 32) {
    const bool more = (kk + 32 < Kd);
    if (more) {   // issue next tile's global loads early (latency hidden)
      av0 = *(const v8h*)pa;    av1 = *(const v8h*)(pa + 8);
      b00 = *(const v8h*)pb0;   b01 = *(const v8h*)(pb0 + 8);
      b10 = *(const v8h*)pb1;   b11 = *(const v8h*)(pb1 + 8);
      pa += 32; pb0 += 32; pb1 += 32;
    }

    Frag fa[4], fb[4];
    #pragma unroll
    for (int i = 0; i < 4; ++i) {
      int r = wm * 64 + i * 16 + lr;
      fa[i].h[0] = *(const v8h*)&As[buf][r][ka];
      fa[i].h[1] = *(const v8h*)&As[buf][r][ka + 8];
    }
    #pragma unroll
    for (int j = 0; j < 4; ++j) {
      int n = wn * 64 + j * 16 + lr;
      fb[j].h[0] = *(const v8h*)&Bs[buf][n][kb];
      fb[j].h[1] = *(const v8h*)&Bs[buf][n][kb + 8];
    }
    #pragma unroll
    for (int i = 0; i < 4; ++i)
      #pragma unroll
      for (int j = 0; j < 4; ++j)
        acc[i][j] = __builtin_amdgcn_wmma_f32_16x16x32_f16(
            false, fa[i].v, false, fb[j].v, (short)0, acc[i][j], false, false);

    if (more) {   // fill the other buffer while this one is being consumed
      *(v8h*)&As[buf ^ 1][srow][soff]           = av0;
      *(v8h*)&As[buf ^ 1][srow][soff + 8]       = av1;
      *(v8h*)&Bs[buf ^ 1][srow][soff]           = b00;
      *(v8h*)&Bs[buf ^ 1][srow][soff + 8]       = b01;
      *(v8h*)&Bs[buf ^ 1][srow + 128][soff]     = b10;
      *(v8h*)&Bs[buf ^ 1][srow + 128][soff + 8] = b11;
    }
    __syncthreads();
    buf ^= 1;
  }

  // epilogue: C layout -> VGPR r holds M = r + (lane>=16 ? 8 : 0), N = lane&15
  const int moff = (lane >> 4) * 8;
  #pragma unroll
  for (int i = 0; i < 4; ++i) {
    #pragma unroll
    for (int j = 0; j < 4; ++j) {
      const int gr0 = row0 + mblock + wm * 64 + i * 16 + moff;
      const int gc  = nbase + wn * 64 + j * 16 + lr;
      #pragma unroll
      for (int r = 0; r < 8; ++r) {
        float v = acc[i][j][r];
        if (do_elu) v = elu1(v);
        Out[(size_t)(gr0 + r) * ldc + gc] = v;
      }
    }
  }
}

// ---------------------------------------------------------------------------
extern "C" void kernel_launch(void* const* d_in, const int* in_sizes, int n_in,
                              void* d_out, int out_size, void* d_ws, size_t ws_size,
                              hipStream_t stream) {
  (void)in_sizes; (void)n_in; (void)out_size; (void)ws_size;
  const float* z   = (const float*)d_in[0];
  const float* c   = (const float*)d_in[1];
  const float* g0w = (const float*)d_in[2];
  const float* g0b = (const float*)d_in[3];
  const float* g1w = (const float*)d_in[4];
  const float* g1b = (const float*)d_in[5];
  const float* g2w = (const float*)d_in[6];
  const float* g2b = (const float*)d_in[7];
  const float* w0  = (const float*)d_in[8];
  const float* b0  = (const float*)d_in[9];
  const float* w1  = (const float*)d_in[10];
  const float* b1  = (const float*)d_in[11];
  const float* w2  = (const float*)d_in[12];
  const float* b2  = (const float*)d_in[13];
  const float* w3  = (const float*)d_in[14];
  const float* b3  = (const float*)d_in[15];
  float* out = (float*)d_out;

  const int KD0 = 8 * 320 + 32;    // 2592
  const int KD1 = 8 * 1088 + 32;   // 8736

  char* ws = (char*)d_ws;
  size_t off = 0;
  auto alloc = [&](size_t bytes) -> void* {
    off = (off + 255) & ~(size_t)255;
    void* p = ws + off;
    off += bytes;
    return p;
  };
  _Float16* W0 = (_Float16*)alloc((size_t)KD0 * 1024 * 2);
  _Float16* W1 = (_Float16*)alloc((size_t)KD1 * 1024 * 2);
  _Float16* W2 = (_Float16*)alloc((size_t)KD1 * 1024 * 2);
  _Float16* W3 = (_Float16*)alloc((size_t)KD1 * 512  * 2);
  float*    coeff     = (float*)alloc((size_t)B_ROWS * 8 * 4);
  float*    layer_out = (float*)alloc((size_t)B_ROWS * 1024 * 4);
  _Float16* Abuf      = (_Float16*)alloc((size_t)CHUNK * KD1 * 2);

  // 1. gating coefficients
  gate_kernel<<<B_ROWS, 128, 0, stream>>>(z, c, g0w, g0b, g1w, g1b, g2w, g2b,
                                          coeff);
  // 2. weight stacking f32 -> f16, transposed to [N][Kd]
  convert_w_t<<<dim3(KD0 / 32, 1024 / 32), 256, 0, stream>>>(w0, b0, W0,
                                                             8 * 320, KD0, 1024);
  convert_w_t<<<dim3(KD1 / 32, 1024 / 32), 256, 0, stream>>>(w1, b1, W1,
                                                             8 * 1088, KD1, 1024);
  convert_w_t<<<dim3(KD1 / 32, 1024 / 32), 256, 0, stream>>>(w2, b2, W2,
                                                             8 * 1088, KD1, 1024);
  convert_w_t<<<dim3(KD1 / 32, 512 / 32),  256, 0, stream>>>(w3, b3, W3,
                                                             8 * 1088, KD1, 512);

  // 3. four layers, chunked over rows (in-place layer_out is safe per-chunk)
  struct Layer {
    const float* prev; int P; int J; int Kd;
    const _Float16* W; int N; float* out; int ldc; int elu;
  } layers[4] = {
    { c,         256,  320,  KD0, W0, 1024, layer_out, 1024, 1 },
    { layer_out, 1024, 1088, KD1, W1, 1024, layer_out, 1024, 1 },
    { layer_out, 1024, 1088, KD1, W2, 1024, layer_out, 1024, 1 },
    { layer_out, 1024, 1088, KD1, W3, 512,  out,       512,  0 },
  };

  for (int l = 0; l < 4; ++l) {
    const Layer& L = layers[l];
    for (int row0 = 0; row0 < B_ROWS; row0 += CHUNK) {
      ln_scale<<<CHUNK, 256, 0, stream>>>(z, L.prev, coeff, Abuf,
                                          row0, L.P, L.J, L.Kd);
      gemm_moe<<<dim3(L.N / 256, CHUNK / 128), 256, 0, stream>>>(
          Abuf, L.W, L.out, L.Kd, L.N, L.ldc, row0, L.elu);
    }
  }
}
